// LANTModelBK_46591805227164
// MI455X (gfx1250) — compile-verified, matrix operand order checked by reading, not run on hardware
//
#include <hip/hip_runtime.h>

typedef __attribute__((ext_vector_type(16))) __bf16 v16bf;
typedef __attribute__((ext_vector_type(8)))  float  v8f;

// ---------- helpers ----------
__device__ __forceinline__ unsigned short f32_to_bf16_rne(float f) {
  unsigned int u = __float_as_uint(f);
  u += 0x7FFFu + ((u >> 16) & 1u);          // round-to-nearest-even
  return (unsigned short)(u >> 16);
}

__device__ __forceinline__ void atomAddF(float* p, float v) {
  unsafeAtomicAdd(p, v);                    // native global_atomic_add_f32
}

__device__ __forceinline__ void atomMaxF(float* p, float v) {
  // IEEE float max via integer atomics (valid with -inf init)
  if (v >= 0.0f) atomicMax((int*)p, __float_as_int(v));
  else           atomicMin((unsigned int*)p, __float_as_uint(v));
}

// ---------- utility kernels ----------
__global__ void zero_f32(float* __restrict__ p, size_t n) {
  size_t i = (size_t)blockIdx.x * blockDim.x + threadIdx.x;
  if (i < n) p[i] = 0.0f;
}

// Convert W[128,128] f32 -> bf16 swizzled into per-lane WMMA B-fragment order:
// out[((ktile*8 + ntile)*32 + lane)*16 + e] = bf16( W[ktile*32 + kmap(e,hi)][ntile*16 + mm] )
__global__ void swizzle_w_bf16(const float* __restrict__ W, unsigned short* __restrict__ out) {
  int t = blockIdx.x * blockDim.x + threadIdx.x;   // 16384
  if (t >= 16384) return;
  int e     = t & 15;
  int lane  = (t >> 4) & 31;
  int ntile = (t >> 9) & 7;
  int ktile = t >> 12;
  int hi = lane >> 4, mm = lane & 15;
  int kk = ktile * 32 + ((e < 8) ? (hi * 8 + e) : (16 + hi * 8 + (e - 8)));
  out[t] = f32_to_bf16_rne(W[kk * 128 + ntile * 16 + mm]);
}

// attw[h*128 + k] = sum_c W[k, h*64+c] * att[h*64+c]  (fold attention vec into lin weight)
__global__ void make_attw(const float* __restrict__ W, const float* __restrict__ att,
                          float* __restrict__ attw) {
  int t = threadIdx.x;              // 256 threads: (k,h)
  int k = t >> 1, h = t & 1;
  float s = 0.0f;
  #pragma unroll 4
  for (int c = 0; c < 64; ++c) s += W[k * 128 + h * 64 + c] * att[h * 64 + c];
  attw[h * 128 + k] = s;
}

// ---------- WMMA GEMM:  Out[N,128] = A[N,128] @ W[128,128] (+bias) ----------
// Wsw is the pre-swizzled bf16 weight (fragment-contiguous per lane).
__global__ void __launch_bounds__(256)
gemm_n128_bf16(const float* __restrict__ A, const unsigned short* __restrict__ Wsw,
               const float* __restrict__ bias, float* __restrict__ Out, int N) {
  __shared__ unsigned short ldsA[16 * 128];    // 4 KB: one 16-row stripe, bf16
  const int tid = threadIdx.x;
  const int rowBase = blockIdx.x * 16;

  // stage A tile: 512 float4 chunks / 256 threads; pack 4xbf16 -> one b64 LDS store
  for (int i = tid; i < 16 * 32; i += 256) {
    int r = i >> 5, c4 = i & 31;
    int row = rowBase + r;
    float4 v = (row < N) ? ((const float4*)A)[(size_t)row * 32 + c4]
                         : make_float4(0.f, 0.f, 0.f, 0.f);
    unsigned long long pk =  (unsigned long long)f32_to_bf16_rne(v.x)
                          | ((unsigned long long)f32_to_bf16_rne(v.y) << 16)
                          | ((unsigned long long)f32_to_bf16_rne(v.z) << 32)
                          | ((unsigned long long)f32_to_bf16_rne(v.w) << 48);
    *(unsigned long long*)&ldsA[r * 128 + c4 * 4] = pk;
  }
  __syncthreads();

  const int wv = tid >> 5;        // 8 waves -> 8 column tiles (full 128 cols)
  const int lane = tid & 31;
  const int hi = lane >> 4;
  const int mm = lane & 15;
  const int n0 = wv * 16;

  v8f acc = {};
  #pragma unroll
  for (int kt = 0; kt < 4; ++kt) {
    union { v16bf v; uint4 q[2]; } af, bf;
    // A: lane's 16 elems are two contiguous 8xbf16 runs -> 2x ds_load_b128
    const unsigned short* abase = &ldsA[mm * 128 + kt * 32 + hi * 8];
    af.q[0] = *(const uint4*)(abase);          // K = kt*32 + hi*8 + 0..7
    af.q[1] = *(const uint4*)(abase + 16);     // K = kt*32 + 16 + hi*8 + 0..7
    // B: swizzled -> 32 contiguous bytes per lane, coalesced across wave
    const uint4* wbase = (const uint4*)(Wsw + ((((kt * 8 + wv) * 32) + lane) << 4));
    bf.q[0] = wbase[0];
    bf.q[1] = wbase[1];
    acc = __builtin_amdgcn_wmma_f32_16x16x32_bf16(false, af.v, false, bf.v,
                                                  (short)0, acc, false, false);
  }

  const float badd = bias ? bias[n0 + mm] : 0.0f;
  #pragma unroll
  for (int v = 0; v < 8; ++v) {               // C/D: VGPR v -> row v + 8*hi, col = lane&15
    int row = rowBase + hi * 8 + v;
    if (row < N) Out[(size_t)row * 128 + n0 + mm] = acc[v] + badd;
  }
}

// ---------- per-node attention scalar: a[n,h] = X[n,:] . attw[h,:] ----------
__global__ void node_att(const float* __restrict__ X, const float* __restrict__ attw,
                         float* __restrict__ a, int N) {
  int t = blockIdx.x * blockDim.x + threadIdx.x;
  if (t >= N * 2) return;
  int n = t >> 1, h = t & 1;
  const float4* x = (const float4*)(X + (size_t)n * 128);
  const float4* w = (const float4*)(attw + h * 128);
  float s = 0.0f;
  #pragma unroll
  for (int k = 0; k < 32; ++k) {
    float4 xv = x[k], wv = w[k];
    s += xv.x * wv.x + xv.y * wv.y + xv.z * wv.z + xv.w * wv.w;
  }
  a[t] = s;
}

// ---------- segment softmax over edges ----------
__global__ void init_softmax(float* __restrict__ emax, float* __restrict__ denom, int n) {
  int t = blockIdx.x * blockDim.x + threadIdx.x;
  if (t < n) { emax[t] = -__builtin_inff(); denom[t] = 0.0f; }
}

__global__ void edge_max(const int* __restrict__ src, const int* __restrict__ dst,
                         const int* __restrict__ permS, const int* __restrict__ permD,
                         const float* __restrict__ a_s, const float* __restrict__ a_d,
                         float* __restrict__ emax, int E) {
  int t = blockIdx.x * blockDim.x + threadIdx.x;
  if (t >= E * 2) return;
  int e = t >> 1, h = t & 1;
  int s = src[e], d = dst[e];
  int sp = permS ? permS[s] : s;
  int dp = permD ? permD[d] : d;
  float v = a_s[sp * 2 + h] + a_d[dp * 2 + h];
  v = (v > 0.0f) ? v : 0.2f * v;              // leaky_relu(0.2)
  atomMaxF(&emax[d * 2 + h], v);
}

__global__ void edge_exp(const int* __restrict__ src, const int* __restrict__ dst,
                         const int* __restrict__ permS, const int* __restrict__ permD,
                         const float* __restrict__ a_s, const float* __restrict__ a_d,
                         const float* __restrict__ emax, float* __restrict__ ee,
                         float* __restrict__ denom, int E) {
  int t = blockIdx.x * blockDim.x + threadIdx.x;
  if (t >= E * 2) return;
  int e = t >> 1, h = t & 1;
  int s = src[e], d = dst[e];
  int sp = permS ? permS[s] : s;
  int dp = permD ? permD[d] : d;
  float v = a_s[sp * 2 + h] + a_d[dp * 2 + h];
  v = (v > 0.0f) ? v : 0.2f * v;
  float x = __expf(v - emax[d * 2 + h]);
  ee[t] = x;
  atomAddF(&denom[d * 2 + h], x);
}

// one wave per edge: lane l handles channels 4l..4l+3 (head = c>>6)
__global__ void __launch_bounds__(256)
edge_agg(const int* __restrict__ src, const int* __restrict__ dst,
         const int* __restrict__ permS, const float* __restrict__ hs,
         const float* __restrict__ ee, const float* __restrict__ denom,
         float* __restrict__ out, int E) {
  int e = blockIdx.x * 8 + (threadIdx.x >> 5);
  if (e >= E) return;
  int lane = threadIdx.x & 31;
  int s = src[e], d = dst[e];
  int sp = permS ? permS[s] : s;
  const float* hrow = hs + (size_t)sp * 128;
  __builtin_prefetch(hrow, 0, 0);             // global_prefetch_b8
  int c0 = lane * 4;
  int h = c0 >> 6;
  float alpha = ee[(size_t)e * 2 + h] / fmaxf(denom[d * 2 + h], 1e-16f);
  float4 m = ((const float4*)hrow)[lane];
  float* o = out + (size_t)d * 128 + c0;
  atomAddF(o + 0, m.x * alpha);
  atomAddF(o + 1, m.y * alpha);
  atomAddF(o + 2, m.z * alpha);
  atomAddF(o + 3, m.w * alpha);
}

// ---------- epilogue: out = prelu(agg + bias); block-reduced column sums ----------
__global__ void __launch_bounds__(256)
bias_prelu(float* __restrict__ out, const float* __restrict__ bias,
           const float* __restrict__ pr, float* __restrict__ sumAcc, int N) {
  __shared__ float red[256];
  int c = threadIdx.x & 127;
  int sub = threadIdx.x >> 7;                 // 0 or 1
  int row0 = blockIdx.x * 32;
  int rowEnd = row0 + 32; if (rowEnd > N) rowEnd = N;
  float b = bias[c], p = pr[c];
  float acc = 0.0f;
  for (int r = row0 + sub; r < rowEnd; r += 2) {
    size_t idx = (size_t)r * 128 + c;
    float v = out[idx] + b;
    v = (v > 0.0f) ? v : p * v;
    out[idx] = v;
    acc += v;
  }
  if (sumAcc) {
    red[threadIdx.x] = acc;
    __syncthreads();
    if (sub == 0) atomAddF(&sumAcc[c], red[c] + red[c + 128]);
  }
}

__global__ void summary_k(const float* __restrict__ sm, const float* __restrict__ st,
                          float* __restrict__ s, float invNM, float invNT) {
  int c = threadIdx.x;
  float v = 0.5f * (sm[c] * invNM + st[c] * invNT);
  s[c] = 1.0f / (1.0f + __expf(-v));
}

// ---------- launcher ----------
extern "C" void kernel_launch(void* const* d_in, const int* in_sizes, int n_in,
                              void* d_out, int out_size, void* d_ws, size_t ws_size,
                              hipStream_t stream) {
  const float* x_member = (const float*)d_in[0];
  const float* x_team   = (const float*)d_in[1];
  const int* mt_src = (const int*)d_in[2];
  const int* mt_dst = (const int*)d_in[3];
  const int* tm_src = (const int*)d_in[4];
  const int* tm_dst = (const int*)d_in[5];
  const int* perm_m = (const int*)d_in[6];
  const int* perm_t = (const int*)d_in[7];
  const float* W_lin_m = (const float*)d_in[8];
  const float* b_lin_m = (const float*)d_in[9];
  const float* W_lin_t = (const float*)d_in[10];
  const float* b_lin_t = (const float*)d_in[11];
  const float* Ws_mt   = (const float*)d_in[12];
  const float* Wd_mt   = (const float*)d_in[13];
  const float* atts_mt = (const float*)d_in[14];
  const float* attd_mt = (const float*)d_in[15];
  const float* bias_mt = (const float*)d_in[16];
  const float* Ws_tm   = (const float*)d_in[17];
  const float* Wd_tm   = (const float*)d_in[18];
  const float* atts_tm = (const float*)d_in[19];
  const float* attd_tm = (const float*)d_in[20];
  const float* bias_tm = (const float*)d_in[21];
  const float* prelu_m = (const float*)d_in[22];
  const float* prelu_t = (const float*)d_in[23];

  const int NM = in_sizes[0] / 128;
  const int NT = in_sizes[1] / 128;
  const int E  = in_sizes[2];
  const int NX = (NM > NT) ? NM : NT;

  // output sections: (pos_m, pos_t, neg_m, neg_t, s)
  float* out   = (float*)d_out;
  float* pos_m = out;
  float* pos_t = pos_m + (size_t)NM * 128;
  float* neg_m = pos_t + (size_t)NT * 128;
  float* neg_t = neg_m + (size_t)NM * 128;
  float* s_out = neg_t + (size_t)NT * 128;

  // workspace carve (256B aligned)
  char* wp = (char*)d_ws;
  auto carve = [&](size_t bytes) { char* p = wp; wp += (bytes + 255) & ~(size_t)255; return p; };
  float* xm     = (float*)carve((size_t)NM * 128 * 4);
  float* xt     = (float*)carve((size_t)NT * 128 * 4);
  float* hs_mt  = (float*)carve((size_t)NM * 128 * 4);
  float* hs_tm  = (float*)carve((size_t)NT * 128 * 4);
  float* a_s_mt = (float*)carve((size_t)NM * 2 * 4);
  float* a_d_mt = (float*)carve((size_t)NT * 2 * 4);
  float* a_s_tm = (float*)carve((size_t)NT * 2 * 4);
  float* a_d_tm = (float*)carve((size_t)NM * 2 * 4);
  float* emax   = (float*)carve((size_t)NX * 2 * 4);
  float* denom  = (float*)carve((size_t)NX * 2 * 4);
  float* ee     = (float*)carve((size_t)E * 2 * 4);
  float* aw_s_mt = (float*)carve(256 * 4);
  float* aw_d_mt = (float*)carve(256 * 4);
  float* aw_s_tm = (float*)carve(256 * 4);
  float* aw_d_tm = (float*)carve(256 * 4);
  float* sum_m   = (float*)carve(128 * 4);
  float* sum_t   = (float*)carve(128 * 4);
  unsigned short* wsw_lin_m = (unsigned short*)carve(16384 * 2);
  unsigned short* wsw_lin_t = (unsigned short*)carve(16384 * 2);
  unsigned short* wsw_s_mt  = (unsigned short*)carve(16384 * 2);
  unsigned short* wsw_s_tm  = (unsigned short*)carve(16384 * 2);
  (void)ws_size; (void)n_in;

  // 1) zero accumulation targets
  {
    size_t n = (size_t)out_size;
    zero_f32<<<(unsigned)((n + 255) / 256), 256, 0, stream>>>(out, n);
    zero_f32<<<1, 256, 0, stream>>>(sum_m, 128);
    zero_f32<<<1, 256, 0, stream>>>(sum_t, 128);
  }

  // 2) swizzled bf16 weights + folded attention vectors
  swizzle_w_bf16<<<64, 256, 0, stream>>>(W_lin_m, wsw_lin_m);
  swizzle_w_bf16<<<64, 256, 0, stream>>>(W_lin_t, wsw_lin_t);
  swizzle_w_bf16<<<64, 256, 0, stream>>>(Ws_mt,   wsw_s_mt);
  swizzle_w_bf16<<<64, 256, 0, stream>>>(Ws_tm,   wsw_s_tm);
  make_attw<<<1, 256, 0, stream>>>(Ws_mt, atts_mt, aw_s_mt);
  make_attw<<<1, 256, 0, stream>>>(Wd_mt, attd_mt, aw_d_mt);
  make_attw<<<1, 256, 0, stream>>>(Ws_tm, atts_tm, aw_s_tm);
  make_attw<<<1, 256, 0, stream>>>(Wd_tm, attd_tm, aw_d_tm);

  // 3) projections + GAT source transforms (WMMA)
  gemm_n128_bf16<<<(NM + 15) / 16, 256, 0, stream>>>(x_member, wsw_lin_m, b_lin_m, xm, NM);
  gemm_n128_bf16<<<(NT + 15) / 16, 256, 0, stream>>>(x_team,   wsw_lin_t, b_lin_t, xt, NT);
  gemm_n128_bf16<<<(NM + 15) / 16, 256, 0, stream>>>(xm, wsw_s_mt, nullptr, hs_mt, NM);
  gemm_n128_bf16<<<(NT + 15) / 16, 256, 0, stream>>>(xt, wsw_s_tm, nullptr, hs_tm, NT);

  // 4) per-node attention scalars (a_s = x @ (Ws.att_s), a_d = x @ (Wd.att_d))
  node_att<<<(NM * 2 + 255) / 256, 256, 0, stream>>>(xm, aw_s_mt, a_s_mt, NM);
  node_att<<<(NT * 2 + 255) / 256, 256, 0, stream>>>(xt, aw_d_mt, a_d_mt, NT);
  node_att<<<(NT * 2 + 255) / 256, 256, 0, stream>>>(xt, aw_s_tm, a_s_tm, NT);
  node_att<<<(NM * 2 + 255) / 256, 256, 0, stream>>>(xm, aw_d_tm, a_d_tm, NM);

  // 5) four aggregations (pos/neg x mt/tm); neg pass reuses hs via perm indirection
  auto run_conv = [&](const int* esrc, const int* edst, const int* pS, const int* pD,
                      const float* a_s, const float* a_d, const float* hs,
                      float* outSec, int Nd, const float* bias, const float* pr,
                      float* sumAcc) {
    int n2 = Nd * 2;
    init_softmax<<<(n2 + 255) / 256, 256, 0, stream>>>(emax, denom, n2);
    edge_max<<<(E * 2 + 255) / 256, 256, 0, stream>>>(esrc, edst, pS, pD, a_s, a_d, emax, E);
    edge_exp<<<(E * 2 + 255) / 256, 256, 0, stream>>>(esrc, edst, pS, pD, a_s, a_d, emax, ee, denom, E);
    edge_agg<<<(E + 7) / 8, 256, 0, stream>>>(esrc, edst, pS, hs, ee, denom, outSec, E);
    bias_prelu<<<(Nd + 31) / 32, 256, 0, stream>>>(outSec, bias, pr, sumAcc, Nd);
  };

  run_conv(mt_src, mt_dst, nullptr, nullptr, a_s_mt, a_d_mt, hs_mt, pos_t, NT, bias_mt, prelu_t, sum_t);
  run_conv(tm_src, tm_dst, nullptr, nullptr, a_s_tm, a_d_tm, hs_tm, pos_m, NM, bias_tm, prelu_m, sum_m);
  run_conv(mt_src, mt_dst, perm_m, perm_t, a_s_mt, a_d_mt, hs_mt, neg_t, NT, bias_mt, prelu_t, nullptr);
  run_conv(tm_src, tm_dst, perm_t, perm_m, a_s_tm, a_d_tm, hs_tm, neg_m, NM, bias_tm, prelu_m, nullptr);

  // 6) summary s = sigmoid(0.5*(mean(pos_m) + mean(pos_t)))
  summary_k<<<1, 128, 0, stream>>>(sum_m, sum_t, s_out, 1.0f / NM, 1.0f / NT);
}
